// HeteroGAT_88510686036238
// MI455X (gfx1250) — compile-verified, matrix operand order checked by reading, not run on hardware
//
#include <hip/hip_runtime.h>

#define NEG_SLOPE 0.2f
#define LDS_STRIDE 136   // 128 + 8 ushort pad: keeps 16B alignment, spreads LDS banks

typedef __attribute__((ext_vector_type(16))) __bf16          v16bf;
typedef __attribute__((ext_vector_type(8)))  float           v8f;
typedef __attribute__((ext_vector_type(8)))  unsigned short  us8;

union Frag {
  unsigned short u[16];
  us8            h[2];
  v16bf          v;
};

__device__ __forceinline__ unsigned short f2bf(float x) {
  unsigned u = __float_as_uint(x);
  unsigned r = u + 0x7FFFu + ((u >> 16) & 1u);   // round-to-nearest-even
  return (unsigned short)(r >> 16);
}

// order-preserving float <-> uint key for atomicMax-based segment max
__device__ __forceinline__ unsigned fkey(float x) {
  unsigned u = __float_as_uint(x);
  return (u & 0x80000000u) ? ~u : (u | 0x80000000u);
}
__device__ __forceinline__ float funkey(unsigned k) {
  unsigned u = (k & 0x80000000u) ? (k & 0x7FFFFFFFu) : ~k;
  return __uint_as_float(u);
}

// -------------------------------------------------------------------------
// Tiled bf16-WMMA GEMM: C[M,128] = act(A[M,128] @ W[128,128] + bias)
// Optional fused per-head attention dot: attn_out[m,h] = sum_d C[m,h*32+d]*avec[h*32+d]
// A is f32 in global, converted to bf16 into LDS; W transposed into LDS.
// Block = 256 threads = 8 waves; wave w computes rows [16w,16w+16) x all 128 cols.
// -------------------------------------------------------------------------
template<bool STORE, bool RELU, bool BIAS, bool ATTN>
__global__ __launch_bounds__(256)
void wmma_gemm128(const float* __restrict__ A, const float* __restrict__ W,
                  const float* __restrict__ bias, const float* __restrict__ avec,
                  float* __restrict__ C, float* __restrict__ attn_out, int M)
{
  __shared__ unsigned short sA[128 * LDS_STRIDE];
  __shared__ unsigned short sW[128 * LDS_STRIDE];   // transposed: sW[n*STRIDE + k]
  __shared__ float sRed[128 * 4];                   // per-row per-head attn partials

  const int    tid = threadIdx.x;
  const size_t m0  = (size_t)blockIdx.x * 128;

  if (ATTN) {
    for (int i = tid; i < 512; i += 256) sRed[i] = 0.f;
  }

  // Load W (128x128 f32, row-major [k][n]) -> bf16 transposed LDS [n][k]
  #pragma unroll
  for (int it = 0; it < 16; ++it) {
    const int linear = (it * 256 + tid) * 4;
    const int k = linear >> 7;
    const int n = linear & 127;
    const float4 w4 = *(const float4*)(W + linear);
    sW[(n + 0) * LDS_STRIDE + k] = f2bf(w4.x);
    sW[(n + 1) * LDS_STRIDE + k] = f2bf(w4.y);
    sW[(n + 2) * LDS_STRIDE + k] = f2bf(w4.z);
    sW[(n + 3) * LDS_STRIDE + k] = f2bf(w4.w);
  }
  // Load A tile (128 rows x 128 k) -> bf16 LDS [r][k]
  #pragma unroll
  for (int it = 0; it < 16; ++it) {
    const int linear = (it * 256 + tid) * 4;
    const int r = linear >> 7;
    const int k = linear & 127;
    float4 a4 = make_float4(0.f, 0.f, 0.f, 0.f);
    if (m0 + (size_t)r < (size_t)M) a4 = *(const float4*)(A + (m0 + r) * 128 + k);
    sA[r * LDS_STRIDE + k + 0] = f2bf(a4.x);
    sA[r * LDS_STRIDE + k + 1] = f2bf(a4.y);
    sA[r * LDS_STRIDE + k + 2] = f2bf(a4.z);
    sA[r * LDS_STRIDE + k + 3] = f2bf(a4.w);
  }
  __syncthreads();

  const int wave = tid >> 5;
  const int lane = tid & 31;
  const int hl   = lane >> 4;    // lane half (selects K sub-block per ISA layout)
  const int lr   = lane & 15;
  const int r0   = wave * 16;

  v8f acc[8];
  const v8f vzero = {0.f, 0.f, 0.f, 0.f, 0.f, 0.f, 0.f, 0.f};
  #pragma unroll
  for (int i = 0; i < 8; ++i) acc[i] = vzero;

  const int arow = r0 + lr;
  #pragma unroll
  for (int kb = 0; kb < 4; ++kb) {
    const int kbase = kb * 32 + hl * 8;
    // A fragment: 16-bit A 16x32 layout — elems 0..7 -> K=kbase..+7, 8..15 -> K=kbase+16..+23
    Frag af;
    const us8* pa = (const us8*)(sA + arow * LDS_STRIDE + kbase);
    af.h[0] = pa[0];
    af.h[1] = pa[2];
    #pragma unroll
    for (int ct = 0; ct < 8; ++ct) {
      const int n = ct * 16 + lr;
      // B fragment = A-layout of B^T; W already transposed in LDS
      Frag bf;
      const us8* pb = (const us8*)(sW + n * LDS_STRIDE + kbase);
      bf.h[0] = pb[0];
      bf.h[1] = pb[2];
      acc[ct] = __builtin_amdgcn_wmma_f32_16x16x32_bf16(
          false, af.v, false, bf.v, (short)0, acc[ct], false, false);
    }
  }

  // Epilogue: C/D layout — vgpr r holds M = r + 8*half, N = lane&15
  #pragma unroll
  for (int ct = 0; ct < 8; ++ct) {
    const int   n  = ct * 16 + lr;
    const float bv = BIAS ? bias[n] : 0.f;
    const float av = ATTN ? avec[n] : 0.f;
    #pragma unroll
    for (int r = 0; r < 8; ++r) {
      const int    mloc = r0 + r + hl * 8;
      const size_t grow = m0 + (size_t)mloc;
      if (grow < (size_t)M) {
        float v = acc[ct][r] + bv;
        if (RELU) v = fmaxf(v, 0.f);
        if (STORE) C[grow * 128 + n] = v;
        if (ATTN) atomicAdd(&sRed[mloc * 4 + (n >> 5)], v * av);   // ds_add_f32
      }
    }
  }
  if (ATTN) {
    __syncthreads();
    if (tid < 128) {
      const size_t grow = m0 + (size_t)tid;
      if (grow < (size_t)M) {
        const float4 v = *(const float4*)&sRed[tid * 4];
        *(float4*)(attn_out + grow * 4) = v;
      }
    }
  }
}

// -------------------------------------------------------------------------
// Edge softmax + aggregation kernels
// -------------------------------------------------------------------------
__global__ void init_max_sum(unsigned* __restrict__ mkey, float* __restrict__ sden, int n)
{
  const int i = blockIdx.x * blockDim.x + threadIdx.x;
  if (i < n) { mkey[i] = 0u; sden[i] = 0.f; }
}

// agg[n, c] = bias[c]  (folds GATConv output bias into the scatter buffer)
__global__ void init_node_bias(float* __restrict__ agg, const float* __restrict__ bias, int total)
{
  const int i = blockIdx.x * blockDim.x + threadIdx.x;
  if (i < total) agg[i] = bias[i & 127];
}

// pass 1: e = leakyrelu(el[src] + er[dst]); store e; segment-max via keyed atomicMax
__global__ void edge_logits_max(const int* __restrict__ src, const int* __restrict__ dst,
                                const float* __restrict__ el, const float* __restrict__ er,
                                float* __restrict__ ebuf, unsigned* __restrict__ mkey, int E)
{
  const int e = blockIdx.x * blockDim.x + threadIdx.x;
  if (e >= E) return;
  const int s = src[e], d = dst[e];
  const float4 l = *(const float4*)(el + (size_t)s * 4);
  const float4 r = *(const float4*)(er + (size_t)d * 4);
  float v[4] = { l.x + r.x, l.y + r.y, l.z + r.z, l.w + r.w };
  #pragma unroll
  for (int h = 0; h < 4; ++h) {
    v[h] = v[h] > 0.f ? v[h] : NEG_SLOPE * v[h];
    atomicMax(mkey + (size_t)d * 4 + h, fkey(v[h]));
  }
  *(float4*)(ebuf + (size_t)e * 4) = make_float4(v[0], v[1], v[2], v[3]);
}

// pass 2: ex = exp(e - max[dst]); overwrite ebuf; segment-sum via f32 atomics
__global__ void edge_exp_sum(const int* __restrict__ dst, float* __restrict__ ebuf,
                             const unsigned* __restrict__ mkey, float* __restrict__ sden, int E)
{
  const int e = blockIdx.x * blockDim.x + threadIdx.x;
  if (e >= E) return;
  const int d = dst[e];
  const float4 v = *(const float4*)(ebuf + (size_t)e * 4);
  float ex[4];
  ex[0] = __expf(v.x - funkey(mkey[(size_t)d * 4 + 0]));
  ex[1] = __expf(v.y - funkey(mkey[(size_t)d * 4 + 1]));
  ex[2] = __expf(v.z - funkey(mkey[(size_t)d * 4 + 2]));
  ex[3] = __expf(v.w - funkey(mkey[(size_t)d * 4 + 3]));
  *(float4*)(ebuf + (size_t)e * 4) = make_float4(ex[0], ex[1], ex[2], ex[3]);
  #pragma unroll
  for (int h = 0; h < 4; ++h) unsafeAtomicAdd(sden + (size_t)d * 4 + h, ex[h]);
}

// pass 3: agg[dst] += (ex/sum[dst]) * fs[src] ; one wave (32 lanes x float4) per edge
__global__ void edge_aggregate(const int* __restrict__ src, const int* __restrict__ dst,
                               const float* __restrict__ ebuf, const float* __restrict__ sden,
                               const float* __restrict__ fs, float* __restrict__ agg, int E)
{
  const int t = blockIdx.x * blockDim.x + threadIdx.x;
  const int e = t >> 5;
  if (e >= E) return;
  const int lane = t & 31;
  const int c0   = lane * 4;      // 4 consecutive feats; head = c0>>5 constant per lane
  const int h    = c0 >> 5;
  const int s = src[e], d = dst[e];
  const float a = ebuf[(size_t)e * 4 + h] / sden[(size_t)d * 4 + h];
  const float4 f = *(const float4*)(fs + (size_t)s * 128 + c0);
  float* p = agg + (size_t)d * 128 + c0;
  unsafeAtomicAdd(p + 0, a * f.x);
  unsafeAtomicAdd(p + 1, a * f.y);
  unsafeAtomicAdd(p + 2, a * f.z);
  unsafeAtomicAdd(p + 3, a * f.w);
}

// -------------------------------------------------------------------------
// Host orchestration
// -------------------------------------------------------------------------
extern "C" void kernel_launch(void* const* d_in, const int* in_sizes, int n_in,
                              void* d_out, int out_size, void* d_ws, size_t ws_size,
                              hipStream_t stream)
{
  const float* x_d      = (const float*)d_in[0];
  const float* x_p      = (const float*)d_in[1];
  const int*   src_dp   = (const int*)d_in[2];
  const int*   dst_dp   = (const int*)d_in[3];
  const int*   src_pd   = (const int*)d_in[4];
  const int*   dst_pd   = (const int*)d_in[5];
  const float* W_in_d   = (const float*)d_in[6];
  const float* b_in_d   = (const float*)d_in[7];
  const float* W_in_p   = (const float*)d_in[8];
  const float* b_in_p   = (const float*)d_in[9];
  const float* W_dp     = (const float*)d_in[10];
  const float* al_dp    = (const float*)d_in[11];
  const float* ar_dp    = (const float*)d_in[12];
  const float* bias_dp  = (const float*)d_in[13];
  const float* W_pd     = (const float*)d_in[14];
  const float* al_pd    = (const float*)d_in[15];
  const float* ar_pd    = (const float*)d_in[16];
  const float* bias_pd  = (const float*)d_in[17];
  const float* W_out_d  = (const float*)d_in[18];
  const float* b_out_d  = (const float*)d_in[19];
  const float* W_out_p  = (const float*)d_in[20];
  const float* b_out_p  = (const float*)d_in[21];

  const int Nd   = in_sizes[0] / 128;
  const int Np   = in_sizes[1] / 128;
  const int E1   = in_sizes[2];
  const int E2   = in_sizes[4];
  const int Nmax = Nd > Np ? Nd : Np;
  const int Emax = E1 > E2 ? E1 : E2;

  // workspace layout (floats)
  float* ws   = (float*)d_ws;
  float* B0   = ws;                                  // Nd*128 : h_d  -> agg_d
  float* B1   = B0 + (size_t)Nd * 128;               // Np*128 : h_p
  float* B2   = B1 + (size_t)Np * 128;               // Nmax*128 : fs_dp -> fs_pd
  float* B3   = B2 + (size_t)Nmax * 128;             // Np*128 : agg_p
  float* ebuf = B3 + (size_t)Np * 128;               // Emax*4 : logits -> exp
  float* el   = ebuf + (size_t)Emax * 4;             // Nmax*4
  float* er   = el + (size_t)Nmax * 4;               // Nmax*4
  unsigned* mkey = (unsigned*)(er + (size_t)Nmax * 4); // Nmax*4
  float* sden = (float*)(mkey + (size_t)Nmax * 4);   // Nmax*4

  float* out_drug = (float*)d_out;
  float* out_prot = out_drug + (size_t)Nd * 128;

  const dim3 blk(256);
  const dim3 gNd((Nd + 127) / 128), gNp((Np + 127) / 128);

  // 1) input projections: h = relu(x @ W_in + b_in)
  wmma_gemm128<true, true, true, false><<<gNd, blk, 0, stream>>>(x_d, W_in_d, b_in_d, nullptr, B0, nullptr, Nd);
  wmma_gemm128<true, true, true, false><<<gNp, blk, 0, stream>>>(x_p, W_in_p, b_in_p, nullptr, B1, nullptr, Np);

  // 2) etype dp (drug -> protein): fs = h_d@W_dp (+el), er fused (fd never stored)
  wmma_gemm128<true,  false, false, true><<<gNd, blk, 0, stream>>>(B0, W_dp, nullptr, al_dp, B2, el, Nd);
  wmma_gemm128<false, false, false, true><<<gNp, blk, 0, stream>>>(B1, W_dp, nullptr, ar_dp, nullptr, er, Np);
  init_max_sum<<<dim3((Np * 4 + 255) / 256), blk, 0, stream>>>(mkey, sden, Np * 4);
  init_node_bias<<<dim3((Np * 128 + 255) / 256), blk, 0, stream>>>(B3, bias_dp, Np * 128);
  edge_logits_max<<<dim3((E1 + 255) / 256), blk, 0, stream>>>(src_dp, dst_dp, el, er, ebuf, mkey, E1);
  edge_exp_sum<<<dim3((E1 + 255) / 256), blk, 0, stream>>>(dst_dp, ebuf, mkey, sden, E1);
  edge_aggregate<<<dim3((E1 * 32 + 255) / 256), blk, 0, stream>>>(src_dp, dst_dp, ebuf, sden, B2, B3, E1);

  // 3) etype pd (protein -> drug)
  wmma_gemm128<true,  false, false, true><<<gNp, blk, 0, stream>>>(B1, W_pd, nullptr, al_pd, B2, el, Np);
  wmma_gemm128<false, false, false, true><<<gNd, blk, 0, stream>>>(B0, W_pd, nullptr, ar_pd, nullptr, er, Nd);
  init_max_sum<<<dim3((Nd * 4 + 255) / 256), blk, 0, stream>>>(mkey, sden, Nd * 4);
  init_node_bias<<<dim3((Nd * 128 + 255) / 256), blk, 0, stream>>>(B0, bias_pd, Nd * 128); // h_d dead now
  edge_logits_max<<<dim3((E2 + 255) / 256), blk, 0, stream>>>(src_pd, dst_pd, el, er, ebuf, mkey, E2);
  edge_exp_sum<<<dim3((E2 + 255) / 256), blk, 0, stream>>>(dst_pd, ebuf, mkey, sden, E2);
  edge_aggregate<<<dim3((E2 * 32 + 255) / 256), blk, 0, stream>>>(src_pd, dst_pd, ebuf, sden, B2, B0, E2);

  // 4) output projections
  wmma_gemm128<true, false, true, false><<<gNd, blk, 0, stream>>>(B0, W_out_d, b_out_d, nullptr, out_drug, nullptr, Nd);
  wmma_gemm128<true, false, true, false><<<gNp, blk, 0, stream>>>(B3, W_out_p, b_out_p, nullptr, out_prot, nullptr, Np);
}
